// SVRNN_Model_68298569941705
// MI455X (gfx1250) — matrix operand order repaired and checked by reading, not compiled, not run on hardware
//
#include <hip/hip_runtime.h>

// ---------------------------------------------------------------------------
// SVRNN (Mamba-ConvLSTM) forward for MI455X / gfx1250.
// GEMM-shaped work (projections, 5x5 convs as implicit GEMM, 1x1 convs) goes
// through v_wmma_f32_16x16x32_bf16. LDS tiles are stored in WMMA *fragment
// order* so each lane fetches its whole operand as 2x ds_load_b128; writers
// pack bf16 pairs (1 b32 store per 2 elements). 64x64 block tiles, 4 waves,
// each wave = 2x2 WMMA tiles with A/B fragment reuse. Tile fills are two-phase
// (batch-issue all raw loads, then convert/pack) so loads are not serialized
// by per-element waits; boundary guards are branchless (clamp + select).
// ---------------------------------------------------------------------------

typedef __attribute__((ext_vector_type(16))) __bf16 v16bf;
typedef __attribute__((ext_vector_type(8)))  float  v8f;

static constexpr int  Bb   = 2;          // batch
static constexpr int  Tt   = 12;         // frames
static constexpr int  Ss   = 1024;       // 32*32 spatial
static constexpr int  NHc  = 64;         // hidden channels
static constexpr int  DMm  = 448;        // 7*NH (mamba d)
static constexpr int  EDd  = 896;        // 2*DM
static constexpr int  DRr  = 28;         // dt rank
static constexpr int  NST  = 16;         // ssm states
static constexpr int  DBC  = 60;         // 28 + 16 + 16
static constexpr int  TOUT = 11;         // output steps

__device__ inline float sigm(float x)     { return 1.f / (1.f + __expf(-x)); }
__device__ inline float siluf(float x)    { return x * sigm(x); }
__device__ inline float softplusf(float x){ return (x > 20.f) ? x : log1pf(__expf(x)); }

__device__ inline unsigned pack_bf16(float a, float b)
{
    union { unsigned u; __bf16 h[2]; } t;
    t.h[0] = (__bf16)a; t.h[1] = (__bf16)b;
    return t.u;
}

// Fragment-order slot decode (shared by A and B writers).
//   pi in [0,1024): sub = pi>>8 (16-wide subtile), r = pi&255,
//   lw = r>>3 (writer lane 0..31), pe = (r&7)*2 (even element index).
// A-fragment (16x32, ISA 7.12.2): lane l holds m=l&15; e<8 -> k=(l>>4)*8+e,
//   e>=8 -> k=16+(l>>4)*8+(e-8).  Pairs (pe,pe+1) <-> (k,k+1), k even.
__device__ inline int a_k_of(int lw, int pe)
{
    int half = lw >> 4;
    return (pe < 8) ? half * 8 + pe : 16 + half * 8 + (pe - 8);
}
// B-fragment (32x16): lane l holds n=l&15; same k order as A-fragment.
__device__ inline int b_k_of(int lw, int pe) { return a_k_of(lw, pe); }

#define WMMA_BF16(A_, B_, C_) \
    __builtin_amdgcn_wmma_f32_16x16x32_bf16(false, (A_), false, (B_), (short)0, (C_), false, false)

// ---------------------------------------------------------------------------
// Generic WMMA GEMM:  C[M,N] = A[M,K] * B   (f32 in/out, bf16 WMMA, f32 acc)
//   A: A[m*lda + k];  B: transB ? B[n*ldb + k] : B[k*ldb + n];  C: C[m*ldc+n]
// grid: (ceil(N/64), ceil(M/64), batch), 128 threads = 4 waves.
// NOTE: lda/ldb must be even when the fast (float2) path can trigger -- true
// for every call site in this file.
// ---------------------------------------------------------------------------
__global__ __launch_bounds__(128)
void svrnn_gemm_wmma(const float* __restrict__ A, const float* __restrict__ B,
                     float* __restrict__ C, int M, int N, int K,
                     int lda, int ldb, int ldc, int transB,
                     long sA, long sB, long sC)
{
    A += (long)blockIdx.z * sA;
    B += (long)blockIdx.z * sB;
    C += (long)blockIdx.z * sC;

    __shared__ v16bf Af[64];   // 4 m-subtiles x 32 lanes, fragment order
    __shared__ v16bf Bf[64];   // 4 n-subtiles x 32 lanes, fragment order
    unsigned* Aw = (unsigned*)Af;
    unsigned* Bw = (unsigned*)Bf;

    const int tid  = threadIdx.x;
    const int wave = tid >> 5;
    const int lane = tid & 31;
    const int m0   = blockIdx.y * 64;
    const int n0   = blockIdx.x * 64;
    const int mi2  = (wave >> 1) * 2;   // first m-subtile of this wave
    const int ni2  = (wave & 1) * 2;    // first n-subtile of this wave

    // Per-thread slot decode is loop-invariant (pi = tid + it*128).
    int slw[8], spe[8], ssub[8];
#pragma unroll
    for (int it = 0; it < 8; ++it) {
        int pi = tid + it * 128;
        ssub[it] = pi >> 8;
        int r = pi & 255;
        slw[it] = r >> 3;
        spe[it] = (r & 7) * 2;
    }

    v8f acc00 = {}, acc01 = {}, acc10 = {}, acc11 = {};

    for (int k0 = 0; k0 < K; k0 += 32) {
        const bool fullA = (m0 + 64 <= M) && (k0 + 32 <= K);
        const bool fullB = (n0 + 64 <= N) && (k0 + 32 <= K);

        // ---- A tile: gather phase then pack phase ----
        if (fullA) {
            if (k0 + 32 < K)
                __builtin_prefetch(A + (long)(m0 + (tid >> 1)) * lda + k0 + 32, 0, 1);
            float2 fa[8];
#pragma unroll
            for (int it = 0; it < 8; ++it) {
                int gm = m0 + ssub[it] * 16 + (slw[it] & 15);
                int gk = k0 + a_k_of(slw[it], spe[it]);
                fa[it] = *(const float2*)(A + (long)gm * lda + gk);
            }
#pragma unroll
            for (int it = 0; it < 8; ++it)
                Aw[tid + it * 128] = pack_bf16(fa[it].x, fa[it].y);
        } else {
            float r0[8], r1[8]; bool o0[8], o1[8];
#pragma unroll
            for (int it = 0; it < 8; ++it) {
                int gm = m0 + ssub[it] * 16 + (slw[it] & 15);
                int gk = k0 + a_k_of(slw[it], spe[it]);
                bool okm = gm < M;
                o0[it] = okm && gk < K;
                o1[it] = okm && gk + 1 < K;
                r0[it] = A[o0[it] ? (long)gm * lda + gk     : 0];
                r1[it] = A[o1[it] ? (long)gm * lda + gk + 1 : 0];
            }
#pragma unroll
            for (int it = 0; it < 8; ++it)
                Aw[tid + it * 128] = pack_bf16(o0[it] ? r0[it] : 0.f,
                                               o1[it] ? r1[it] : 0.f);
        }
        // ---- B tile: gather phase then pack phase ----
        if (fullB && transB) {
            if (k0 + 32 < K)
                __builtin_prefetch(B + (long)(n0 + (tid >> 1)) * ldb + k0 + 32, 0, 1);
            float2 fb[8];
#pragma unroll
            for (int it = 0; it < 8; ++it) {
                int gn = n0 + ssub[it] * 16 + (slw[it] & 15);
                int gk = k0 + b_k_of(slw[it], spe[it]);
                fb[it] = *(const float2*)(B + (long)gn * ldb + gk);
            }
#pragma unroll
            for (int it = 0; it < 8; ++it)
                Bw[tid + it * 128] = pack_bf16(fb[it].x, fb[it].y);
        } else if (fullB) {
            float r0[8], r1[8];
#pragma unroll
            for (int it = 0; it < 8; ++it) {
                int gn = n0 + ssub[it] * 16 + (slw[it] & 15);
                int gk = k0 + b_k_of(slw[it], spe[it]);
                r0[it] = B[(long)gk * ldb + gn];
                r1[it] = B[(long)(gk + 1) * ldb + gn];
            }
#pragma unroll
            for (int it = 0; it < 8; ++it)
                Bw[tid + it * 128] = pack_bf16(r0[it], r1[it]);
        } else {
            float r0[8], r1[8]; bool o0[8], o1[8];
#pragma unroll
            for (int it = 0; it < 8; ++it) {
                int gn = n0 + ssub[it] * 16 + (slw[it] & 15);
                int gk = k0 + b_k_of(slw[it], spe[it]);
                bool okn = gn < N;
                o0[it] = okn && gk < K;
                o1[it] = okn && gk + 1 < K;
                long p0 = transB ? (long)gn * ldb + gk     : (long)gk * ldb + gn;
                long p1 = transB ? (long)gn * ldb + gk + 1 : (long)(gk + 1) * ldb + gn;
                r0[it] = B[o0[it] ? p0 : 0];
                r1[it] = B[o1[it] ? p1 : 0];
            }
#pragma unroll
            for (int it = 0; it < 8; ++it)
                Bw[tid + it * 128] = pack_bf16(o0[it] ? r0[it] : 0.f,
                                               o1[it] ? r1[it] : 0.f);
        }
        __syncthreads();

        v16bf a0 = Af[(mi2 + 0) * 32 + lane];
        v16bf a1 = Af[(mi2 + 1) * 32 + lane];
        v16bf b0 = Bf[(ni2 + 0) * 32 + lane];
        v16bf b1 = Bf[(ni2 + 1) * 32 + lane];
        acc00 = WMMA_BF16(a0, b0, acc00);
        acc01 = WMMA_BF16(a0, b1, acc01);
        acc10 = WMMA_BF16(a1, b0, acc10);
        acc11 = WMMA_BF16(a1, b1, acc11);
        __syncthreads();
    }

    const int nlo = lane & 15;
    const int mhi = (lane >> 4) * 8;
#pragma unroll
    for (int i = 0; i < 2; ++i) {
#pragma unroll
        for (int j = 0; j < 2; ++j) {
            const v8f& ac = (i == 0) ? (j == 0 ? acc00 : acc01)
                                     : (j == 0 ? acc10 : acc11);
            int nn = n0 + (ni2 + j) * 16 + nlo;
            if (nn < N) {
#pragma unroll
                for (int r = 0; r < 8; ++r) {
                    int mm = m0 + (mi2 + i) * 16 + mhi + r;
                    if (mm < M) C[(long)mm * ldc + nn] = ac[r];
                }
            }
        }
    }
}

// ---------------------------------------------------------------------------
// 5x5 SAME conv as implicit GEMM: Y[b,co,p] = sum_{ci,ky,kx} W[co,ci,ky,kx]*X
// M = Cout, N = 1024 spatial, K = Cin*25.  grid: (16, ceil(Cout/64), B).
// Same fragment-order LDS scheme; B tile is a two-phase branchless im2col
// gather (issue all 16 loads, then select/convert/pack).
// ---------------------------------------------------------------------------
__global__ __launch_bounds__(128)
void svrnn_conv5x5(const float* __restrict__ Wt, const float* __restrict__ X,
                   float* __restrict__ Y, int Cout, int Cin,
                   long xBatch, long yBatch)
{
    const int K = Cin * 25;
    const float* Xb = X + (long)blockIdx.z * xBatch;
    float*       Yb = Y + (long)blockIdx.z * yBatch;

    __shared__ v16bf Af[64];
    __shared__ v16bf Bf[64];
    unsigned* Aw = (unsigned*)Af;
    unsigned* Bw = (unsigned*)Bf;

    const int tid  = threadIdx.x;
    const int wave = tid >> 5;
    const int lane = tid & 31;
    const int m0   = blockIdx.y * 64;
    const int n0   = blockIdx.x * 64;
    const int mi2  = (wave >> 1) * 2;
    const int ni2  = (wave & 1) * 2;

    int slw[8], spe[8], ssub[8];
#pragma unroll
    for (int it = 0; it < 8; ++it) {
        int pi = tid + it * 128;
        ssub[it] = pi >> 8;
        int r = pi & 255;
        slw[it] = r >> 3;
        spe[it] = (r & 7) * 2;
    }

    v8f acc00 = {}, acc01 = {}, acc10 = {}, acc11 = {};

    for (int k0 = 0; k0 < K; k0 += 32) {
        const bool fullA = (m0 + 64 <= Cout) && (k0 + 32 <= K);
        // ---- A tile: weights, row-major [Cout, K] ----
        if (fullA) {
            float r0[8], r1[8];
#pragma unroll
            for (int it = 0; it < 8; ++it) {
                int gm = m0 + ssub[it] * 16 + (slw[it] & 15);
                int gk = k0 + a_k_of(slw[it], spe[it]);
                r0[it] = Wt[(long)gm * K + gk];
                r1[it] = Wt[(long)gm * K + gk + 1];
            }
#pragma unroll
            for (int it = 0; it < 8; ++it)
                Aw[tid + it * 128] = pack_bf16(r0[it], r1[it]);
        } else {
            float r0[8], r1[8]; bool o0[8], o1[8];
#pragma unroll
            for (int it = 0; it < 8; ++it) {
                int gm = m0 + ssub[it] * 16 + (slw[it] & 15);
                int gk = k0 + a_k_of(slw[it], spe[it]);
                bool okm = gm < Cout;
                o0[it] = okm && gk < K;
                o1[it] = okm && gk + 1 < K;
                r0[it] = Wt[o0[it] ? (long)gm * K + gk     : 0];
                r1[it] = Wt[o1[it] ? (long)gm * K + gk + 1 : 0];
            }
#pragma unroll
            for (int it = 0; it < 8; ++it)
                Aw[tid + it * 128] = pack_bf16(o0[it] ? r0[it] : 0.f,
                                               o1[it] ? r1[it] : 0.f);
        }
        // ---- B tile: two-phase branchless im2col gather ----
        {
            float r0[8], r1[8]; bool o0[8], o1[8];
#pragma unroll
            for (int it = 0; it < 8; ++it) {
                int n  = n0 + ssub[it] * 16 + (slw[it] & 15);  // spatial pos < 1024
                int py0 = n >> 5, px0 = n & 31;
                int gk0 = k0 + b_k_of(slw[it], spe[it]);
#pragma unroll
                for (int u = 0; u < 2; ++u) {
                    int gk = gk0 + u;
                    int ci = gk / 25, rr = gk - ci * 25;
                    int ky = rr / 5, kx = rr - ky * 5;
                    int py = py0 + ky - 2, px = px0 + kx - 2;
                    bool ok = (gk < K) & (py >= 0) & (py < 32) & (px >= 0) & (px < 32);
                    float v = Xb[ok ? (long)ci * Ss + py * 32 + px : 0];
                    if (u == 0) { o0[it] = ok; r0[it] = v; }
                    else        { o1[it] = ok; r1[it] = v; }
                }
            }
#pragma unroll
            for (int it = 0; it < 8; ++it)
                Bw[tid + it * 128] = pack_bf16(o0[it] ? r0[it] : 0.f,
                                               o1[it] ? r1[it] : 0.f);
        }
        __syncthreads();

        v16bf a0 = Af[(mi2 + 0) * 32 + lane];
        v16bf a1 = Af[(mi2 + 1) * 32 + lane];
        v16bf b0 = Bf[(ni2 + 0) * 32 + lane];
        v16bf b1 = Bf[(ni2 + 1) * 32 + lane];
        acc00 = WMMA_BF16(a0, b0, acc00);
        acc01 = WMMA_BF16(a0, b1, acc01);
        acc10 = WMMA_BF16(a1, b0, acc10);
        acc11 = WMMA_BF16(a1, b1, acc11);
        __syncthreads();
    }

    const int nlo = lane & 15;
    const int mhi = (lane >> 4) * 8;
#pragma unroll
    for (int i = 0; i < 2; ++i) {
#pragma unroll
        for (int j = 0; j < 2; ++j) {
            const v8f& ac = (i == 0) ? (j == 0 ? acc00 : acc01)
                                     : (j == 0 ? acc10 : acc11);
            int nn = n0 + (ni2 + j) * 16 + nlo;
#pragma unroll
            for (int r = 0; r < 8; ++r) {
                int mm = m0 + (mi2 + i) * 16 + mhi + r;
                if (mm < Cout) Yb[(long)mm * Ss + nn] = ac[r];
            }
        }
    }
}

// --------------------------- elementwise kernels ---------------------------

// x[b,C,1024] -> y[b,1024,C]
__global__ void svrnn_cp2pc(const float* __restrict__ x, float* __restrict__ y, int C)
{
    long i = (long)blockIdx.x * blockDim.x + threadIdx.x;
    long total = (long)Bb * Ss * C;
    if (i >= total) return;
    int c = (int)(i % C); long t = i / C;
    int p = (int)(t % Ss); int b = (int)(t / Ss);
    y[i] = x[((long)b * C + c) * Ss + p];
}

// x[b,1024,C] -> y[b,C,1024]
__global__ void svrnn_pc2cp(const float* __restrict__ x, float* __restrict__ y, int C)
{
    long i = (long)blockIdx.x * blockDim.x + threadIdx.x;
    long total = (long)Bb * Ss * C;
    if (i >= total) return;
    int p = (int)(i % Ss); long t = i / Ss;
    int c = (int)(t % C); int b = (int)(t / C);
    y[i] = x[((long)b * Ss + p) * C + c];
}

__global__ void svrnn_add(float* __restrict__ a, const float* __restrict__ b, long total)
{
    long i = (long)blockIdx.x * blockDim.x + threadIdx.x;
    if (i < total) a[i] += b[i];
}

// RMSNorm over last dim; one 128-thread block per row
__global__ __launch_bounds__(128)
void svrnn_rmsnorm(const float* __restrict__ x, const float* __restrict__ w,
                   float* __restrict__ y, int d)
{
    long row = blockIdx.x;
    const float* xr = x + row * (long)d;
    float*       yr = y + row * (long)d;
    __shared__ float red[128];
    int tid = threadIdx.x;
    float s = 0.f;
    for (int i = tid; i < d; i += 128) { float v = xr[i]; s += v * v; }
    red[tid] = s; __syncthreads();
    for (int o = 64; o > 0; o >>= 1) {
        if (tid < o) red[tid] += red[tid + o];
        __syncthreads();
    }
    float inv = rsqrtf(red[0] / (float)d + 1e-5f);
    for (int i = tid; i < d; i += 128) yr[i] = xr[i] * inv * w[i];
}

// LayerNorm over whole (C,H,W) volume per sample, per-element affine; in-place.
__global__ __launch_bounds__(256)
void svrnn_ln3(float* __restrict__ x, const float* __restrict__ w,
               const float* __restrict__ bm, long n)
{
    x += (long)blockIdx.x * n;
    __shared__ float r1[256], r2[256];
    int tid = threadIdx.x;
    float s = 0.f, q = 0.f;
    for (long i = tid; i < n; i += 256) { float v = x[i]; s += v; q += v * v; }
    r1[tid] = s; r2[tid] = q; __syncthreads();
    for (int o = 128; o > 0; o >>= 1) {
        if (tid < o) { r1[tid] += r1[tid + o]; r2[tid] += r2[tid + o]; }
        __syncthreads();
    }
    float mu  = r1[0] / (float)n;
    float var = r2[0] / (float)n - mu * mu;
    float inv = rsqrtf(var + 1e-5f);
    for (long i = tid; i < n; i += 256) x[i] = (x[i] - mu) * inv * w[i] + bm[i];
}

// Depthwise causal conv1d (D_CONV=4) + bias + SiLU; reads xin from xz rows.
__global__ void svrnn_dwconv(const float* __restrict__ xz, const float* __restrict__ w,
                             const float* __restrict__ bias, float* __restrict__ xc)
{
    long i = (long)blockIdx.x * blockDim.x + threadIdx.x;
    const long total = (long)Bb * Ss * EDd;
    if (i >= total) return;
    int  e  = (int)(i % EDd);
    long bl = i / EDd;
    int  l  = (int)(bl % Ss);
    long b  = bl / Ss;
    const float* base = xz + (b * Ss) * (long)(2 * EDd) + e;  // xin[b, :, e]
    float r[4];
#pragma unroll
    for (int j = 0; j < 4; ++j) {
        int ls = l + j - 3;
        r[j] = base[ls >= 0 ? (long)ls * (2 * EDd) : 0];
    }
    float acc = bias[e];
#pragma unroll
    for (int j = 0; j < 4; ++j) {
        int ls = l + j - 3;
        acc += (ls >= 0 ? r[j] : 0.f) * w[e * 4 + j];
    }
    xc[i] = siluf(acc);
}

// Fused selective-scan: per (b,e) thread keeps 16 states in registers; Bm/Cm
// rows broadcast through LDS each step.
__global__ __launch_bounds__(128)
void svrnn_scan(const float* __restrict__ dpre, const float* __restrict__ dtb,
                const float* __restrict__ Alog, const float* __restrict__ xc,
                const float* __restrict__ dbc, const float* __restrict__ Dp,
                float* __restrict__ y)
{
    int e = blockIdx.x * 128 + threadIdx.x;
    int b = blockIdx.y;
    __shared__ float Bs[NST], Cs[NST];
    float A[NST], h[NST];
    float bias = 0.f, dv = 0.f;
#pragma unroll
    for (int n = 0; n < NST; ++n) { h[n] = 0.f; A[n] = 0.f; }
    if (e < EDd) {
        bias = dtb[e]; dv = Dp[e];
#pragma unroll
        for (int n = 0; n < NST; ++n) A[n] = -__expf(Alog[e * NST + n]);
    }
    for (int l = 0; l < Ss; ++l) {
        long row = (long)b * Ss + l;
        if (threadIdx.x < NST)
            Bs[threadIdx.x] = dbc[row * DBC + DRr + threadIdx.x];
        else if (threadIdx.x < 2 * NST)
            Cs[threadIdx.x - NST] = dbc[row * DBC + DRr + NST + (threadIdx.x - NST)];
        __syncthreads();
        if (e < EDd) {
            float delta = softplusf(dpre[row * EDd + e] + bias);
            float xv    = xc[row * EDd + e];
            float acc   = 0.f;
#pragma unroll
            for (int n = 0; n < NST; ++n) {
                float da = __expf(delta * A[n]);
                h[n] = da * h[n] + delta * Bs[n] * xv;
                acc += h[n] * Cs[n];
            }
            y[row * EDd + e] = acc + dv * xv;
        }
        __syncthreads();
    }
}

// y *= silu(z)   (z = second half of xz row)
__global__ void svrnn_ygate(float* __restrict__ y, const float* __restrict__ xz)
{
    long i = (long)blockIdx.x * blockDim.x + threadIdx.x;
    const long total = (long)Bb * Ss * EDd;
    if (i >= total) return;
    int  e   = (int)(i % EDd);
    long row = i / EDd;
    float z = xz[row * (long)(2 * EDd) + EDd + e];
    y[i] *= siluf(z);
}

// i/f/g gates -> new c, new m, and mem = concat(c_new, m_new)
__global__ void svrnn_gates(const float* __restrict__ xcat, const float* __restrict__ hcat,
                            const float* __restrict__ mcat, float* __restrict__ c,
                            float* __restrict__ m, float* __restrict__ mem)
{
    int i = blockIdx.x * blockDim.x + threadIdx.x;
    if (i >= Bb * NHc * Ss) return;
    int p  = i & (Ss - 1);
    int t  = i >> 10;
    int ch = t % NHc;
    int b  = t / NHc;
    long xb = (long)b * DMm * Ss;
    long hb = (long)b * 4 * NHc * Ss;
    long mb = (long)b * 3 * NHc * Ss;
    auto X  = [&](int g) { return xcat[xb + ((long)(g * NHc + ch)) * Ss + p]; };
    auto Hv = [&](int g) { return hcat[hb + ((long)(g * NHc + ch)) * Ss + p]; };
    auto Mv = [&](int g) { return mcat[mb + ((long)(g * NHc + ch)) * Ss + p]; };

    float it = sigm(X(0) + Hv(0));
    float ft = sigm(X(1) + Hv(1) + 1.f);
    float gt = tanhf(X(2) + Hv(2));
    float cn = ft * c[i] + it * gt;
    float itp = sigm(X(3) + Mv(0));
    float ftp = sigm(X(4) + Mv(1) + 1.f);
    float gtp = tanhf(X(5) + Mv(2));
    float mn = ftp * m[i] + itp * gtp;
    c[i] = cn; m[i] = mn;
    mem[((long)b * 2 * NHc + ch) * Ss + p]       = cn;
    mem[((long)b * 2 * NHc + NHc + ch) * Ss + p] = mn;
}

// h_new = sigmoid(o_x + o_h + ln(Wo*mem)) * tanh(Wlast*mem)
__global__ void svrnn_outgate(const float* __restrict__ xcat, const float* __restrict__ hcat,
                              const float* __restrict__ wo, const float* __restrict__ wl,
                              float* __restrict__ h)
{
    int i = blockIdx.x * blockDim.x + threadIdx.x;
    if (i >= Bb * NHc * Ss) return;
    int p  = i & (Ss - 1);
    int t  = i >> 10;
    int ch = t % NHc;
    int b  = t / NHc;
    float ox = xcat[((long)b * DMm + 6 * NHc + ch) * Ss + p];
    float oh = hcat[((long)b * 4 * NHc + 3 * NHc + ch) * Ss + p];
    h[i] = sigm(ox + oh + wo[i]) * tanhf(wl[i]);
}

// write step output into d_out (B, 11, 32, 32, 1) and autoregressive buffer
__global__ void svrnn_copyout(const float* __restrict__ wout, float* __restrict__ dout,
                              float* __restrict__ last, int t)
{
    int i = blockIdx.x * blockDim.x + threadIdx.x;
    if (i >= Bb * Ss) return;
    int p = i & (Ss - 1);
    int b = i >> 10;
    float v = wout[i];
    dout[((long)(b * TOUT + t)) * Ss + p] = v;
    last[i] = v;
}

// ---------------------------------------------------------------------------
// Host orchestration
// ---------------------------------------------------------------------------
extern "C" void kernel_launch(void* const* d_in, const int* in_sizes, int n_in,
                              void* d_out, int out_size, void* d_ws, size_t ws_size,
                              hipStream_t stream)
{
    (void)in_sizes; (void)n_in; (void)out_size; (void)ws_size;

    // JAX pytree leaf order (dict keys sorted):
    // 0: frames, 1: W_out, then per cell (23 leaves each, cell0 then cell1):
    //   Wh, Wlast, Wm, Wo, Wx, lnh_b, lnh_w, lnm_b, lnm_w, lno_b, lno_w,
    //   lnx_b, lnx_w, A_log, D, conv_b, conv_w, dt_bias, dt_proj, in_proj,
    //   norm_w, out_proj, x_proj
    const float* frames = (const float*)d_in[0];
    const float* W_out  = (const float*)d_in[1];

    struct CellP {
        const float *Wh, *Wlast, *Wm, *Wo, *Wx;
        const float *lnh_b, *lnh_w, *lnm_b, *lnm_w, *lno_b, *lno_w, *lnx_b, *lnx_w;
        const float *A_log, *Dp, *conv_b, *conv_w, *dt_bias, *dt_proj, *in_proj;
        const float *norm_w, *out_proj, *x_proj;
        int Cin;
    };
    CellP cp[2];
    for (int c = 0; c < 2; ++c) {
        int o = 2 + c * 23;
        cp[c].Wh      = (const float*)d_in[o + 0];
        cp[c].Wlast   = (const float*)d_in[o + 1];
        cp[c].Wm      = (const float*)d_in[o + 2];
        cp[c].Wo      = (const float*)d_in[o + 3];
        cp[c].Wx      = (const float*)d_in[o + 4];
        cp[c].lnh_b   = (const float*)d_in[o + 5];
        cp[c].lnh_w   = (const float*)d_in[o + 6];
        cp[c].lnm_b   = (const float*)d_in[o + 7];
        cp[c].lnm_w   = (const float*)d_in[o + 8];
        cp[c].lno_b   = (const float*)d_in[o + 9];
        cp[c].lno_w   = (const float*)d_in[o + 10];
        cp[c].lnx_b   = (const float*)d_in[o + 11];
        cp[c].lnx_w   = (const float*)d_in[o + 12];
        cp[c].A_log   = (const float*)d_in[o + 13];
        cp[c].Dp      = (const float*)d_in[o + 14];
        cp[c].conv_b  = (const float*)d_in[o + 15];
        cp[c].conv_w  = (const float*)d_in[o + 16];
        cp[c].dt_bias = (const float*)d_in[o + 17];
        cp[c].dt_proj = (const float*)d_in[o + 18];
        cp[c].in_proj = (const float*)d_in[o + 19];
        cp[c].norm_w  = (const float*)d_in[o + 20];
        cp[c].out_proj= (const float*)d_in[o + 21];
        cp[c].x_proj  = (const float*)d_in[o + 22];
        cp[c].Cin     = (c == 0) ? 1 : NHc;
    }

    // ---- workspace carve (floats) ----
    float* wp = (float*)d_ws;
    auto take = [&](size_t n) { float* r = wp; wp += n; return r; };
    float* h0    = take((size_t)Bb * NHc * Ss);
    float* h1    = take((size_t)Bb * NHc * Ss);
    float* c0    = take((size_t)Bb * NHc * Ss);
    float* c1    = take((size_t)Bb * NHc * Ss);
    float* mst   = take((size_t)Bb * NHc * Ss);
    float* last  = take((size_t)Bb * Ss);
    float* xcat  = take((size_t)Bb * DMm * Ss);
    float* sbuf  = take((size_t)Bb * Ss * DMm);
    float* xn    = take((size_t)Bb * Ss * DMm);
    float* xz    = take((size_t)Bb * Ss * 2 * EDd);
    float* xc    = take((size_t)Bb * Ss * EDd);
    float* dbc   = take((size_t)Bb * Ss * DBC);
    float* dpre  = take((size_t)Bb * Ss * EDd);
    float* ybuf  = take((size_t)Bb * Ss * EDd);
    float* proj  = take((size_t)Bb * Ss * DMm);
    float* hcat  = take((size_t)Bb * 4 * NHc * Ss);
    float* mcat  = take((size_t)Bb * 3 * NHc * Ss);
    float* mem   = take((size_t)Bb * 2 * NHc * Ss);
    float* wo    = take((size_t)Bb * NHc * Ss);
    float* wl    = take((size_t)Bb * NHc * Ss);
    float* wout  = take((size_t)Bb * Ss);

    const size_t stBytes = (size_t)Bb * NHc * Ss * sizeof(float);
    hipMemsetAsync(h0, 0, stBytes, stream);
    hipMemsetAsync(h1, 0, stBytes, stream);
    hipMemsetAsync(c0, 0, stBytes, stream);
    hipMemsetAsync(c1, 0, stBytes, stream);
    hipMemsetAsync(mst, 0, stBytes, stream);

    auto gemm = [&](const float* A, const float* Bm, float* C, int M, int N, int K,
                    int lda, int ldb, int ldc, int tB, long sA, long sB, long sC, int bat) {
        dim3 g((unsigned)((N + 63) / 64), (unsigned)((M + 63) / 64), (unsigned)bat);
        svrnn_gemm_wmma<<<g, 128, 0, stream>>>(A, Bm, C, M, N, K, lda, ldb, ldc, tB, sA, sB, sC);
    };
    auto ew = [&](long total) { return dim3((unsigned)((total + 255) / 256)); };

    auto run_cell = [&](const CellP& P, const float* inp, long inBatch,
                        float* hst, float* cst) {
        const long BS = (long)Bb * Ss;
        // 1) xcat = conv5x5(inp, Wx)
        {
            dim3 g(16u, (unsigned)((DMm + 63) / 64), (unsigned)Bb);
            svrnn_conv5x5<<<g, 128, 0, stream>>>(P.Wx, inp, xcat, DMm, P.Cin,
                                                 inBatch, (long)DMm * Ss);
        }
        // 2) mamba over spatial sequence
        svrnn_cp2pc<<<ew(BS * DMm), 256, 0, stream>>>(xcat, sbuf, DMm);
        svrnn_rmsnorm<<<(unsigned)BS, 128, 0, stream>>>(sbuf, P.norm_w, xn, DMm);
        gemm(xn, P.in_proj, xz, (int)BS, 2 * EDd, DMm, DMm, DMm, 2 * EDd, 1, 0, 0, 0, 1);
        svrnn_dwconv<<<ew(BS * EDd), 256, 0, stream>>>(xz, P.conv_w, P.conv_b, xc);
        gemm(xc, P.x_proj, dbc, (int)BS, DBC, EDd, EDd, EDd, DBC, 1, 0, 0, 0, 1);
        gemm(dbc, P.dt_proj, dpre, (int)BS, EDd, DRr, DBC, DRr, EDd, 1, 0, 0, 0, 1);
        svrnn_scan<<<dim3((unsigned)(EDd / 128), (unsigned)Bb), 128, 0, stream>>>(
            dpre, P.dt_bias, P.A_log, xc, dbc, P.Dp, ybuf);
        svrnn_ygate<<<ew(BS * EDd), 256, 0, stream>>>(ybuf, xz);
        gemm(ybuf, P.out_proj, proj, (int)BS, DMm, EDd, EDd, EDd, DMm, 1, 0, 0, 0, 1);
        svrnn_add<<<ew(BS * DMm), 256, 0, stream>>>(sbuf, proj, BS * DMm);
        svrnn_pc2cp<<<ew(BS * DMm), 256, 0, stream>>>(sbuf, xcat, DMm);
        // 3) layer norms + branch convs
        svrnn_ln3<<<(unsigned)Bb, 256, 0, stream>>>(xcat, P.lnx_w, P.lnx_b, (long)DMm * Ss);
        {
            dim3 g(16u, (unsigned)((4 * NHc + 63) / 64), (unsigned)Bb);
            svrnn_conv5x5<<<g, 128, 0, stream>>>(P.Wh, hst, hcat, 4 * NHc, NHc,
                                                 (long)NHc * Ss, (long)4 * NHc * Ss);
        }
        svrnn_ln3<<<(unsigned)Bb, 256, 0, stream>>>(hcat, P.lnh_w, P.lnh_b, (long)4 * NHc * Ss);
        {
            dim3 g(16u, (unsigned)((3 * NHc + 63) / 64), (unsigned)Bb);
            svrnn_conv5x5<<<g, 128, 0, stream>>>(P.Wm, mst, mcat, 3 * NHc, NHc,
                                                 (long)NHc * Ss, (long)3 * NHc * Ss);
        }
        svrnn_ln3<<<(unsigned)Bb, 256, 0, stream>>>(mcat, P.lnm_w, P.lnm_b, (long)3 * NHc * Ss);
        // 4) gates -> c_new, m_new, mem
        svrnn_gates<<<ew((long)Bb * NHc * Ss), 256, 0, stream>>>(xcat, hcat, mcat,
                                                                 cst, mst, mem);
        // 5) output gate path
        {
            dim3 g(16u, (unsigned)((NHc + 63) / 64), (unsigned)Bb);
            svrnn_conv5x5<<<g, 128, 0, stream>>>(P.Wo, mem, wo, NHc, 2 * NHc,
                                                 (long)2 * NHc * Ss, (long)NHc * Ss);
        }
        svrnn_ln3<<<(unsigned)Bb, 256, 0, stream>>>(wo, P.lno_w, P.lno_b, (long)NHc * Ss);
        // Wlast 1x1 conv as GEMM, batched over B
        gemm(P.Wlast, mem, wl, NHc, Ss, 2 * NHc, 2 * NHc, Ss, Ss, 0,
             0, (long)2 * NHc * Ss, (long)NHc * Ss, Bb);
        svrnn_outgate<<<ew((long)Bb * NHc * Ss), 256, 0, stream>>>(xcat, hcat, wo, wl, hst);
    };

    // ---- time loop: steps 0..10 ----
    for (int s = 0; s < TOUT; ++s) {
        const float* inp; long inBatch;
        if (s <= 4)      { inp = frames + (long)s * Ss; inBatch = (long)Tt * Ss; }
        else if (s == 5) { inp = frames + (long)6 * Ss; inBatch = (long)Tt * Ss; }
        else             { inp = last;                  inBatch = (long)Ss; }

        run_cell(cp[0], inp, inBatch, h0, c0);
        run_cell(cp[1], h0, (long)NHc * Ss, h1, c1);

        // W_out 1x1 conv: [1,64] x h1[b,64,1024]
        gemm(W_out, h1, wout, 1, Ss, NHc, NHc, Ss, Ss, 0,
             0, (long)NHc * Ss, (long)Ss, Bb);
        svrnn_copyout<<<ew((long)Bb * Ss), 256, 0, stream>>>(wout, (float*)d_out, last, s);
    }
}
